// myModel_6347961663803
// MI455X (gfx1250) — matrix operand order repaired and checked by reading, not compiled, not run on hardware
//
#include <hip/hip_runtime.h>
#include <hip/hip_bf16.h>

typedef __attribute__((ext_vector_type(16))) _Float16 v16h;
typedef __attribute__((ext_vector_type(8)))  float    v8f;

#define D      32
#define R      256
#define GN     1000
#define GPAD   1008   // 63 tiles of 16
#define TSTEPS 4

// Branch-free SELU: scale*max(x,0) + (scale*alpha)*(exp(min(x,0)) - 1).
// Exact: for x >= 0 the exp term is exp(0)-1 == 0.
__device__ __forceinline__ float selu_f(float x) {
    const float scale  = 1.0507009873554805f;
    const float salpha = 1.7580993408473766f;   // scale * alpha
    const float xp = fmaxf(x, 0.0f);
    const float xn = fminf(x, 0.0f);
    return fmaf(scale, xp, salpha * (__expf(xn) - 1.0f));
}
__device__ __forceinline__ float sigmoid_f(float x) {
    return 1.0f / (1.0f + __expf(-x));
}

__device__ __forceinline__ v8f wmma_f16(v16h a, v16h b, v8f c) {
    // (neg_a, A, neg_b, B, c_mod, C, reuse_a, reuse_b)
    return __builtin_amdgcn_wmma_f32_16x16x32_f16(false, a, false, b, (short)0, c,
                                                  false, false);
}

// A fragment of a 16x32 K-slice; this lane holds row m = lane%16.
// elements 0-7  -> K = kbase + hi*8 + 0..7
// elements 8-15 -> K = kbase + 16 + hi*8 + 0..7      (hi = lane/16)
__device__ __forceinline__ v16h load_a_frag(const float* rowp, int kbase, int hi) {
    const float* p0 = rowp + kbase + hi * 8;
    const float* p1 = p0 + 16;
    float4 q0 = *(const float4*)(p0);
    float4 q1 = *(const float4*)(p0 + 4);
    float4 q2 = *(const float4*)(p1);
    float4 q3 = *(const float4*)(p1 + 4);
    v16h a;
    a[0]  = (_Float16)q0.x; a[1]  = (_Float16)q0.y; a[2]  = (_Float16)q0.z; a[3]  = (_Float16)q0.w;
    a[4]  = (_Float16)q1.x; a[5]  = (_Float16)q1.y; a[6]  = (_Float16)q1.z; a[7]  = (_Float16)q1.w;
    a[8]  = (_Float16)q2.x; a[9]  = (_Float16)q2.y; a[10] = (_Float16)q2.z; a[11] = (_Float16)q2.w;
    a[12] = (_Float16)q3.x; a[13] = (_Float16)q3.y; a[14] = (_Float16)q3.z; a[15] = (_Float16)q3.w;
    return a;
}

__device__ __forceinline__ v16h zero_a_frag() {
    v16h a;
#pragma unroll
    for (int i = 0; i < 16; ++i) a[i] = (_Float16)0.0f;
    return a;
}

// B fragment (32x16 slice of row-major W[K x ld]); lane holds column `col`.
// element i -> K = kbase + hi*8 + i (i<8), kbase + 16 + hi*8 + (i-8) (i>=8)
__device__ __forceinline__ v16h load_b_frag(const float* W, int ld, int kbase, int col, int hi) {
    v16h b;
#pragma unroll
    for (int i = 0; i < 16; ++i)
        b[i] = (_Float16)W[(size_t)(kbase + (i < 8 ? hi * 8 + i : 16 + hi * 8 + (i - 8))) * ld + col];
    return b;
}

__global__ void zero_kernel(float4* p, int n4) {
    int i = blockIdx.x * blockDim.x + threadIdx.x;
    int stride = gridDim.x * blockDim.x;
    float4 z = make_float4(0.f, 0.f, 0.f, 0.f);
    for (; i < n4; i += stride) p[i] = z;
}

// ---------------------------------------------------------------------------
// Fused: gather 16-edge tile -> (16x64)@(64x32) WMMA -> +bias -> SELU ->
// atomic scatter-add into agg[second[e]] (segment_sum). agg stays L2-resident.
// ---------------------------------------------------------------------------
__global__ void __launch_bounds__(256)
edge_msg_kernel(const float* __restrict__ ls,
                const int*   __restrict__ efirst,
                const int*   __restrict__ esecond,
                const float* __restrict__ Wmsg,   // 64 x 32
                const float* __restrict__ bmsg,   // 32
                float*       __restrict__ agg,    // N x 32 (zeroed)
                int E) {
    const int lane = threadIdx.x & 31;
    const int hi   = lane >> 4;
    const int c    = lane & 15;
    const int wave   = (blockIdx.x * blockDim.x + threadIdx.x) >> 5;
    const int nwaves = (gridDim.x * blockDim.x) >> 5;

    // Weight fragments, resident across the whole tile loop.
    v16h B00 = load_b_frag(Wmsg, D, 0,  c,      hi);   // K 0-31  (main), N 0-15
    v16h B01 = load_b_frag(Wmsg, D, 0,  16 + c, hi);   // K 0-31  (main), N 16-31
    v16h B10 = load_b_frag(Wmsg, D, D,  c,      hi);   // K 32-63 (neigh), N 0-15
    v16h B11 = load_b_frag(Wmsg, D, D,  16 + c, hi);   // K 32-63 (neigh), N 16-31
    const float bias0 = bmsg[c];
    const float bias1 = bmsg[16 + c];

    const int ntiles = (E + 15) >> 4;
    for (int t = wave; t < ntiles; t += nwaves) {
        const int base = t << 4;
        const bool full = (base + 16 <= E);      // uniform across the wave

        // Prefetch next tile's index cachelines (only streaming HBM traffic here).
        const int nbase = base + (nwaves << 4);
        if (nbase < E) {
            __builtin_prefetch(&efirst[nbase + lane], 0, 0);
            __builtin_prefetch(&esecond[nbase + lane], 0, 0);
        }

        v16h aM, aN;
        if (full) {
            const int erow = base + c;           // A-row m = lane%16
            const int fi = efirst[erow];
            const int si = esecond[erow];
            aM = load_a_frag(ls + (size_t)fi * D, 0, hi);
            aN = load_a_frag(ls + (size_t)si * D, 0, hi);
        } else {
            const int erow = base + c;
            if (erow < E) {
                const int fi = efirst[erow];
                const int si = esecond[erow];
                aM = load_a_frag(ls + (size_t)fi * D, 0, hi);
                aN = load_a_frag(ls + (size_t)si * D, 0, hi);
            } else {
                aM = zero_a_frag();
                aN = zero_a_frag();
            }
        }

        v8f c0 = {};
        v8f c1 = {};
        c0 = wmma_f16(aM, B00, c0);
        c0 = wmma_f16(aN, B10, c0);
        c1 = wmma_f16(aM, B01, c1);
        c1 = wmma_f16(aN, B11, c1);

        if (full) {
#pragma unroll
            for (int v = 0; v < 8; ++v) {
                const int e2 = base + hi * 8 + v;   // C-row m = hi*8 + v
                const int dst = esecond[e2];
                atomicAdd(&agg[(size_t)dst * D + c],      selu_f(c0[v] + bias0));
                atomicAdd(&agg[(size_t)dst * D + 16 + c], selu_f(c1[v] + bias1));
            }
        } else {
#pragma unroll
            for (int v = 0; v < 8; ++v) {
                const int e2 = base + hi * 8 + v;
                if (e2 < E) {
                    const int dst = esecond[e2];
                    atomicAdd(&agg[(size_t)dst * D + c],      selu_f(c0[v] + bias0));
                    atomicAdd(&agg[(size_t)dst * D + 16 + c], selu_f(c1[v] + bias1));
                }
            }
        }
    }
}

// ---------------------------------------------------------------------------
// Fused GRU cell: xi = agg@K + b0 ; hr = h@RK + b1 ; gates ; in-place update.
// 12 WMMAs per 16-node tile (6 N-tiles x {kernel, rec_kernel}).
// ---------------------------------------------------------------------------
__global__ void __launch_bounds__(256)
gru_kernel(float*       __restrict__ ls,    // N x 32 (in/out)
           const float* __restrict__ agg,   // N x 32
           const float* __restrict__ K,     // 32 x 96
           const float* __restrict__ RK,    // 32 x 96
           const float* __restrict__ bias,  // 2 x 96
           int N) {
    const int lane = threadIdx.x & 31;
    const int hi   = lane >> 4;
    const int c    = lane & 15;
    const int wave   = (blockIdx.x * blockDim.x + threadIdx.x) >> 5;
    const int nwaves = (gridDim.x * blockDim.x) >> 5;

    v16h bx[6], bh[6];
#pragma unroll
    for (int n = 0; n < 6; ++n) {
        bx[n] = load_b_frag(K,  96, 0, n * 16 + c, hi);
        bh[n] = load_b_frag(RK, 96, 0, n * 16 + c, hi);
    }

    const int ntiles = (N + 15) >> 4;
    for (int t = wave; t < ntiles; t += nwaves) {
        const int base = t << 4;
        const bool full = (base + 16 <= N);      // uniform across the wave
        const int arow = base + c;

        v16h ax, ah;
        if (full || arow < N) {
            ax = load_a_frag(agg + (size_t)arow * D, 0, hi);
            ah = load_a_frag(ls  + (size_t)arow * D, 0, hi);
        } else {
            ax = zero_a_frag();
            ah = zero_a_frag();
        }

        v8f cx[6], ch[6];
#pragma unroll
        for (int n = 0; n < 6; ++n) {
            v8f z0 = {};
            v8f z1 = {};
            cx[n] = wmma_f16(ax, bx[n], z0);
            ch[n] = wmma_f16(ah, bh[n], z1);
        }
#pragma unroll
        for (int p = 0; p < 2; ++p) {
            const int j   = p * 16 + c;          // output column 0..31
            const float b0z = bias[j],        b1z = bias[96 + j];
            const float b0r = bias[32 + j],   b1r = bias[128 + j];
            const float b0h = bias[64 + j],   b1h = bias[160 + j];
#pragma unroll
            for (int v = 0; v < 8; ++v) {
                const int node = base + hi * 8 + v;
                if (full || node < N) {
                    const float z  = sigmoid_f(cx[p][v]     + b0z + ch[p][v]     + b1z);
                    const float r  = sigmoid_f(cx[2 + p][v] + b0r + ch[2 + p][v] + b1r);
                    const float hc = tanhf(cx[4 + p][v] + b0h + r * (ch[4 + p][v] + b1h));
                    float* hp = &ls[(size_t)node * D + j];
                    const float hold = *hp;
                    *hp = z * hold + (1.0f - z) * hc;
                }
            }
        }
    }
}

__global__ void graph_sum_kernel(const float* __restrict__ ls,
                                 const int*   __restrict__ gids,
                                 float*       __restrict__ gs,
                                 int N) {
    const int idx  = blockIdx.x * blockDim.x + threadIdx.x;
    const int node = idx >> 5;
    const int col  = idx & 31;
    if (node < N) atomicAdd(&gs[(size_t)gids[node] * D + col], ls[(size_t)node * D + col]);
}

// h1 = selu(gs @ W1 + b1)   (GPAD x 32) @ (32 x 256)
__global__ void mlp1_kernel(const float* __restrict__ gs,
                            const float* __restrict__ W1,
                            const float* __restrict__ b1,
                            float*       __restrict__ h1) {
    const int lane = threadIdx.x & 31;
    const int hi   = lane >> 4;
    const int c    = lane & 15;
    const int wave = (blockIdx.x * blockDim.x + threadIdx.x) >> 5;
    const int mt = wave >> 4;
    const int nt = wave & 15;
    if (mt >= GPAD / 16) return;
    const int col = nt * 16 + c;
    v16h a = load_a_frag(gs + (size_t)(mt * 16 + c) * D, 0, hi);
    v16h b = load_b_frag(W1, R, 0, col, hi);
    v8f acc = {};
    acc = wmma_f16(a, b, acc);
    const float bb = b1[col];
#pragma unroll
    for (int v = 0; v < 8; ++v) {
        const int row = mt * 16 + hi * 8 + v;
        h1[(size_t)row * R + col] = selu_f(acc[v] + bb);
    }
}

// h2 = selu(h1 @ W2 + b2)   (GPAD x 256) @ (256 x 256), 8 K-blocks
__global__ void mlp2_kernel(const float* __restrict__ h1,
                            const float* __restrict__ W2,
                            const float* __restrict__ b2,
                            float*       __restrict__ h2) {
    const int lane = threadIdx.x & 31;
    const int hi   = lane >> 4;
    const int c    = lane & 15;
    const int wave = (blockIdx.x * blockDim.x + threadIdx.x) >> 5;
    const int mt = wave >> 4;
    const int nt = wave & 15;
    if (mt >= GPAD / 16) return;
    const int col = nt * 16 + c;
    const float* arow = h1 + (size_t)(mt * 16 + c) * R;
    v8f acc = {};
#pragma unroll
    for (int kb = 0; kb < R / 32; ++kb) {
        v16h a = load_a_frag(arow, kb * 32, hi);
        v16h b = load_b_frag(W2, R, kb * 32, col, hi);
        acc = wmma_f16(a, b, acc);
    }
    const float bb = b2[col];
#pragma unroll
    for (int v = 0; v < 8; ++v) {
        const int row = mt * 16 + hi * 8 + v;
        h2[(size_t)row * R + col] = selu_f(acc[v] + bb);
    }
}

// out = h2 @ W3 + b3   (1000 x 256) @ (256 x 1)
__global__ void mlp3_kernel(const float* __restrict__ h2,
                            const float* __restrict__ W3,
                            const float* __restrict__ b3,
                            float*       __restrict__ out,
                            int G) {
    const int g = blockIdx.x * blockDim.x + threadIdx.x;
    if (g >= G) return;
    const float* row = h2 + (size_t)g * R;
    float acc = b3[0];
#pragma unroll 8
    for (int i = 0; i < R; ++i) acc += row[i] * W3[i];
    out[g] = acc;
}

extern "C" void kernel_launch(void* const* d_in, const int* in_sizes, int n_in,
                              void* d_out, int out_size, void* d_ws, size_t ws_size,
                              hipStream_t stream) {
    const float* states_action = (const float*)d_in[0];
    const int*   graph_ids     = (const int*)  d_in[1];
    const int*   efirst        = (const int*)  d_in[2];
    const int*   esecond       = (const int*)  d_in[3];
    const float* Wmsg          = (const float*)d_in[5];
    const float* bmsg          = (const float*)d_in[6];
    const float* gruK          = (const float*)d_in[7];
    const float* gruRK         = (const float*)d_in[8];
    const float* gruB          = (const float*)d_in[9];
    const float* W1            = (const float*)d_in[10];
    const float* b1            = (const float*)d_in[11];
    const float* W2            = (const float*)d_in[12];
    const float* b2            = (const float*)d_in[13];
    const float* W3            = (const float*)d_in[14];
    const float* b3            = (const float*)d_in[15];
    (void)n_in; (void)out_size; (void)ws_size;

    const int N = in_sizes[0] / D;
    const int E = in_sizes[2];

    char* ws = (char*)d_ws;
    size_t off = 0;
    auto wsalloc = [&](size_t bytes) -> void* {
        void* p = ws + off;
        off = (off + bytes + 255) & ~(size_t)255;
        return p;
    };
    float* ls  = (float*)wsalloc((size_t)N * D * sizeof(float));     // 32 MB
    float* agg = (float*)wsalloc((size_t)N * D * sizeof(float));     // 32 MB
    float* gs  = (float*)wsalloc((size_t)GPAD * D * sizeof(float));
    float* h1  = (float*)wsalloc((size_t)GPAD * R * sizeof(float));
    float* h2  = (float*)wsalloc((size_t)GPAD * R * sizeof(float));

    hipMemcpyAsync(ls, states_action, (size_t)N * D * sizeof(float),
                   hipMemcpyDeviceToDevice, stream);

    for (int step = 0; step < TSTEPS; ++step) {
        zero_kernel<<<2048, 256, 0, stream>>>((float4*)agg, N * D / 4);
        edge_msg_kernel<<<2048, 256, 0, stream>>>(ls, efirst, esecond, Wmsg, bmsg, agg, E);
        gru_kernel<<<1024, 256, 0, stream>>>(ls, agg, gruK, gruRK, gruB, N);
    }

    zero_kernel<<<32, 256, 0, stream>>>((float4*)gs, GPAD * D / 4);
    graph_sum_kernel<<<(N * D + 255) / 256, 256, 0, stream>>>(ls, graph_ids, gs, N);
    mlp1_kernel<<<126, 256, 0, stream>>>(gs, W1, b1, h1);   // 1008 waves = 63x16 tiles
    mlp2_kernel<<<126, 256, 0, stream>>>(h1, W2, b2, h2);
    mlp3_kernel<<<(GN + 255) / 256, 256, 0, stream>>>(h2, W3, b3, (float*)d_out, GN);
}
